// BinaryConv_73375221285495
// MI455X (gfx1250) — compile-verified, matrix operand order checked by reading, not run on hardware
//
#include <hip/hip_runtime.h>

typedef _Float16 v8h  __attribute__((ext_vector_type(8)));
typedef _Float16 v16h __attribute__((ext_vector_type(16)));
typedef float    v8f  __attribute__((ext_vector_type(8)));

#define HN 56
#define WN 56
#define CI 256
#define CO 256
#define BLOCK_M 128
#define LDS_STRIDE 40   // halves per row = 80 bytes (16B aligned, 20-dword pitch)
#define NSTEPS 72       // 3*3 taps * 8 chunks of K=32

#if defined(__HIP_DEVICE_COMPILE__) && defined(__gfx1250__) && \
    __has_builtin(__builtin_amdgcn_global_load_async_to_lds_b128)
#define USE_ASYNC_B 1
#else
#define USE_ASYNC_B 0
#endif

#if USE_ASYNC_B
typedef int v4i __attribute__((vector_size(16)));
typedef __attribute__((address_space(1))) v4i gv4i_t;   // global int4
typedef __attribute__((address_space(3))) v4i lv4i_t;   // LDS int4

__device__ __forceinline__ void async_copy_b128(const void* g, void* l) {
  // global -> LDS, 16B per lane, tracked by ASYNCcnt
  __builtin_amdgcn_global_load_async_to_lds_b128(
      (gv4i_t*)(unsigned long long)g, (lv4i_t*)l, 0, 0);
}

__device__ __forceinline__ void wait_asynccnt0() {
#if __has_builtin(__builtin_amdgcn_s_wait_asynccnt)
  __builtin_amdgcn_s_wait_asynccnt(0);
#else
  asm volatile("s_wait_asynccnt 0x0" ::: "memory");
#endif
}
#endif

// ---------------------------------------------------------------------------
// Kernel 1: binarize weights (sign, sign(0)=0) and transpose
//   w : [kh][kw][ci][co] f32  ->  wt : [kh][kw][co][ci] f16
// ---------------------------------------------------------------------------
__global__ void binarize_transpose_w(const float* __restrict__ w,
                                     _Float16* __restrict__ wt) {
  int idx = blockIdx.x * blockDim.x + threadIdx.x;   // over 9*256*256
  if (idx >= 9 * 256 * 256) return;
  int ci = idx & 255;
  int co = (idx >> 8) & 255;
  int t  = idx >> 16;                                // kh*3+kw
  float v = w[((size_t)t * 256 + ci) * 256 + co];
  float s = (v > 0.0f) ? 1.0f : ((v < 0.0f) ? -1.0f : 0.0f);
  wt[idx] = (_Float16)s;
}

// ---------------------------------------------------------------------------
// Kernel 2: implicit-GEMM binary conv, v_wmma_f32_16x16x32_f16.
// Block: 128 pixels x 256 Cout, 256 threads (8 wave32, 2(M) x 4(N) waves,
// each wave computes a 64x64 tile = 16 WMMA accumulators).
// Double-buffered LDS, one barrier per K-step. A staged via VGPRs (needs
// f32->f16 convert); B staged via GLOBAL_LOAD_ASYNC_TO_LDS_B128 (pure copy),
// drained with s_wait_asynccnt 0 before the barrier.
// ---------------------------------------------------------------------------
__global__ __launch_bounds__(256) void bconv_wmma(
    const float* __restrict__ x, const _Float16* __restrict__ wt,
    float* __restrict__ out) {
  __shared__ __align__(16) _Float16 As[2][BLOCK_M * LDS_STRIDE];  // 2 x 10 KB
  __shared__ __align__(16) _Float16 Bs[2][CO * LDS_STRIDE];       // 2 x 20 KB

  const int tid  = threadIdx.x;
  const int lane = tid & 31;
  const int wave = tid >> 5;
  const int wm   = wave & 1;    // 0..1 : 64 output rows each
  const int wn   = wave >> 1;   // 0..3 : 64 output cols each
  const int lsel = lane >> 4;   // selects K half-runs per ISA 16-bit layout
  const int lrow = lane & 15;   // M (A) / N (B,C,D) within a 16x16 tile

  const int p0 = blockIdx.x * BLOCK_M;

  // A-loader role: 2 threads per pixel row, 16 channels (32B of f16) each
  const int arow  = tid >> 1;          // 0..127
  const int acoff = (tid & 1) * 16;    // halve offset within K=32 chunk
  const int p   = p0 + arow;
  const int bb  = p / (HN * WN);
  const int rem = p - bb * (HN * WN);
  const int hh  = rem / WN;
  const int iw0 = rem - hh * WN;
  const float* xpix =
      x + (size_t)((bb * HN + hh) * WN + iw0) * CI + acoff;
  // B-loader role: thread t stages weight row co=t (32 halves = 64B)
  const _Float16* wrow = wt + (size_t)tid * 256;

  float4 ga[4];   // 16 x-values for A staging
#if !USE_ASYNC_B
  float4 gb[4];   // 64B of f16 weights for B staging (sync fallback)
#endif

  v8f acc[4][4];
#pragma unroll
  for (int i = 0; i < 4; ++i)
#pragma unroll
    for (int j = 0; j < 4; ++j)
#pragma unroll
      for (int r = 0; r < 8; ++r) acc[i][j][r] = 0.0f;

  auto stage_load = [&](int s) {
    const int t9 = s >> 3;          // kh*3+kw
    const int kc = s & 7;
    const int kh = t9 / 3;
    const int kw = t9 - kh * 3;
    const int ih = hh + kh - 1;
    const int iw = iw0 + kw - 1;
    const bool valid = ((unsigned)ih < (unsigned)HN) &&
                       ((unsigned)iw < (unsigned)WN);
    const float* src = xpix + ((kh - 1) * WN + (kw - 1)) * CI + kc * 32;
    if (valid) {
      ga[0] = *(const float4*)(src);
      ga[1] = *(const float4*)(src + 4);
      ga[2] = *(const float4*)(src + 8);
      ga[3] = *(const float4*)(src + 12);
    } else {
      const float4 z = {0.0f, 0.0f, 0.0f, 0.0f};
      ga[0] = z; ga[1] = z; ga[2] = z; ga[3] = z;
    }
#if !USE_ASYNC_B
    const float4* ws = (const float4*)(wrow + (size_t)t9 * 65536 + kc * 32);
    gb[0] = ws[0]; gb[1] = ws[1]; gb[2] = ws[2]; gb[3] = ws[3];
#endif
  };

#if USE_ASYNC_B
  auto async_load_B = [&](int s, int buf) {
    const int t9 = s >> 3;
    const int kc = s & 7;
    const _Float16* src = wrow + (size_t)t9 * 65536 + kc * 32;
    _Float16* dst = &Bs[buf][tid * LDS_STRIDE];
    async_copy_b128(src,      dst);
    async_copy_b128(src + 8,  dst + 8);
    async_copy_b128(src + 16, dst + 16);
    async_copy_b128(src + 24, dst + 24);
  };
#endif

  auto stage_store = [&](int buf) {
    v8h h0, h1;
    h0[0] = (_Float16)ga[0].x; h0[1] = (_Float16)ga[0].y;
    h0[2] = (_Float16)ga[0].z; h0[3] = (_Float16)ga[0].w;
    h0[4] = (_Float16)ga[1].x; h0[5] = (_Float16)ga[1].y;
    h0[6] = (_Float16)ga[1].z; h0[7] = (_Float16)ga[1].w;
    h1[0] = (_Float16)ga[2].x; h1[1] = (_Float16)ga[2].y;
    h1[2] = (_Float16)ga[2].z; h1[3] = (_Float16)ga[2].w;
    h1[4] = (_Float16)ga[3].x; h1[5] = (_Float16)ga[3].y;
    h1[6] = (_Float16)ga[3].z; h1[7] = (_Float16)ga[3].w;
    *(v8h*)(&As[buf][arow * LDS_STRIDE + acoff])     = h0;
    *(v8h*)(&As[buf][arow * LDS_STRIDE + acoff + 8]) = h1;
#if !USE_ASYNC_B
    float4* bd = (float4*)(&Bs[buf][tid * LDS_STRIDE]);
    bd[0] = gb[0]; bd[1] = gb[1]; bd[2] = gb[2]; bd[3] = gb[3];
#endif
  };

  stage_load(0);
#if USE_ASYNC_B
  async_load_B(0, 0);
#endif
  int buf = 0;
  for (int s = 0; s < NSTEPS; ++s) {
    stage_store(buf);
#if USE_ASYNC_B
    wait_asynccnt0();                    // drain this wave's B copy for buf
#endif
    __syncthreads();                     // buf fully staged for all waves
    if (s + 1 < NSTEPS) {
      stage_load(s + 1);                 // overlaps WMMA burst
#if USE_ASYNC_B
      async_load_B(s + 1, buf ^ 1);      // DMA overlaps WMMA burst
#endif
    }

    v16h af[4], bf[4];
#pragma unroll
    for (int i = 0; i < 4; ++i) {
      const int m = wm * 64 + i * 16 + lrow;
      v8h lo = *(v8h*)(&As[buf][m * LDS_STRIDE + lsel * 8]);
      v8h hi = *(v8h*)(&As[buf][m * LDS_STRIDE + 16 + lsel * 8]);
      af[i] = __builtin_shufflevector(lo, hi, 0, 1, 2, 3, 4, 5, 6, 7, 8, 9,
                                      10, 11, 12, 13, 14, 15);
    }
#pragma unroll
    for (int j = 0; j < 4; ++j) {
      const int n = wn * 64 + j * 16 + lrow;
      v8h lo = *(v8h*)(&Bs[buf][n * LDS_STRIDE + lsel * 8]);
      v8h hi = *(v8h*)(&Bs[buf][n * LDS_STRIDE + 16 + lsel * 8]);
      bf[j] = __builtin_shufflevector(lo, hi, 0, 1, 2, 3, 4, 5, 6, 7, 8, 9,
                                      10, 11, 12, 13, 14, 15);
    }
#pragma unroll
    for (int i = 0; i < 4; ++i)
#pragma unroll
      for (int j = 0; j < 4; ++j)
        acc[i][j] = __builtin_amdgcn_wmma_f32_16x16x32_f16(
            false, af[i], false, bf[j], (short)0, acc[i][j], false, false);

    buf ^= 1;
  }

  // ---- store D tiles: lane = N col, VGPR r = M row (+8 for lanes >= 16) ----
#pragma unroll
  for (int i = 0; i < 4; ++i) {
#pragma unroll
    for (int j = 0; j < 4; ++j) {
      const int pbase = p0 + wm * 64 + i * 16 + lsel * 8;
      const int co    = wn * 64 + j * 16 + lrow;
#pragma unroll
      for (int r = 0; r < 8; ++r) {
        out[(size_t)(pbase + r) * CO + co] = acc[i][j][r];
      }
    }
  }
}

// ---------------------------------------------------------------------------
extern "C" void kernel_launch(void* const* d_in, const int* in_sizes, int n_in,
                              void* d_out, int out_size, void* d_ws,
                              size_t ws_size, hipStream_t stream) {
  const float* x = (const float*)d_in[0];
  const float* w = (const float*)d_in[1];
  float* out = (float*)d_out;
  _Float16* wt = (_Float16*)d_ws;  // needs 9*256*256*2 = 1.15 MB scratch

  binarize_transpose_w<<<(9 * 256 * 256 + 255) / 256, 256, 0, stream>>>(w, wt);

  const int n_pixels = 32 * HN * WN;              // 100352, exact /128
  bconv_wmma<<<n_pixels / BLOCK_M, 256, 0, stream>>>(x, wt, out);
}
